// Attention_87591563034992
// MI455X (gfx1250) — matrix operand order repaired and checked by reading, not compiled, not run on hardware
//
#include <hip/hip_runtime.h>
#include <hip/hip_bf16.h>

// ---------------- problem constants (from reference) ----------------
constexpr int B_   = 2;
constexpr int S_   = 2048;
constexpr int HID_ = 2048;
constexpr int NH_  = 16;
constexpr int HD_  = 128;          // HID / NH
constexpr float SCALE_ = 0.08838834764831845f; // 1/sqrt(HD)

// ---------------- CDNA5 WMMA types ----------------
typedef __attribute__((ext_vector_type(16))) __bf16 v16bf;
typedef __attribute__((ext_vector_type(8)))  float  v8f;
typedef __attribute__((ext_vector_type(4)))  unsigned int v4u;

union Frag { v16bf v; v4u u[2]; };

__device__ __forceinline__ v8f wmma_bf16(v16bf a, v16bf b, v8f c) {
    // (neg_a, A, neg_b, B, c_mod, C, reuse_a, reuse_b)
    return __builtin_amdgcn_wmma_f32_16x16x32_bf16(false, a, false, b,
                                                   (short)0, c, false, false);
}

__device__ __forceinline__ __bf16 f2bf(float f) {
    unsigned u = __builtin_bit_cast(unsigned, f);
    unsigned r = u + 0x7FFFu + ((u >> 16) & 1u);     // round-to-nearest-even
    unsigned short h = (unsigned short)(r >> 16);
    return __builtin_bit_cast(__bf16, h);
}
__device__ __forceinline__ float bf2f(__bf16 x) {
    unsigned short h = __builtin_bit_cast(unsigned short, x);
    unsigned u = ((unsigned)h) << 16;
    return __builtin_bit_cast(float, u);
}

// ---------------- CDNA5 async copy: global -> LDS (ASYNCcnt) ----------------
// Per-lane 16B copy, no VGPR round trip. dst-first / addr-second / "off" operand
// order matches the vflat syntax seen in gfx1250 disassembly.
__device__ __forceinline__ void async_copy_b128(__bf16* lds_dst, const __bf16* gsrc) {
    unsigned loff = (unsigned)(size_t)lds_dst;       // LDS address = low 32 bits of the pointer
    asm volatile("global_load_async_to_lds_b128 %0, %1, off"
                 :: "v"(loff), "v"(gsrc) : "memory");
}
template<int N>
__device__ __forceinline__ void wait_async() {
    asm volatile("s_wait_asynccnt %0" :: "i"(N) : "memory");
}

// A fragment, 16x32 bf16 (MxK): lane&15 = row, lane-half picks K octets {k..k+7, k+16..k+23} / {+8,+24}
__device__ __forceinline__ Frag load_frag_a(const __bf16* __restrict__ base, int ld,
                                            int row, int k, int lane) {
    const int half = lane >> 4, ln = lane & 15;
    const __bf16* p = base + (size_t)(row + ln) * ld + k + (half << 3);
    Frag f;
    f.u[0] = *(const v4u*)p;
    f.u[1] = *(const v4u*)(p + 16);
    return f;
}
// B fragment, 32x16 bf16 (KxN): lane&15 = column, lane-half picks K octets {k..k+7, k+16..k+23} / {+8,+24}
__device__ __forceinline__ Frag load_frag_b(const __bf16* __restrict__ base, int ld,
                                            int col, int k, int lane) {
    const int half = lane >> 4, ln = lane & 15;
    const __bf16* p = base + (size_t)(col + ln) * ld + k + (half << 3);
    Frag f;
    f.u[0] = *(const v4u*)p;
    f.u[1] = *(const v4u*)(p + 16);
    return f;
}

// ---------------- fp32 -> bf16 conversion ----------------
__global__ __launch_bounds__(256) void cvt_f32_bf16(const float* __restrict__ in,
                                                    __bf16* __restrict__ out, int n) {
    int i = (blockIdx.x * 256 + threadIdx.x) * 4;
    if (i < n) {
        float4 v = *(const float4*)(in + i);
        out[i + 0] = f2bf(v.x);
        out[i + 1] = f2bf(v.y);
        out[i + 2] = f2bf(v.z);
        out[i + 3] = f2bf(v.w);
    }
}

// ---------------- GEMM: C(MxN) = A(MxK) * B(NxK)^T, bf16 in, f32 acc ----------------
// Block = 256 threads = 8 waves; wave computes a 16x64 tile. The shared 64x32 B tile
// is staged to LDS with double-buffered async copies (ASYNCcnt); A stays per-wave
// direct global b128 loads (private rows) + prefetch of the next K strip.
template<bool OUT_F32>
__global__ __launch_bounds__(256) void gemm_bf16_nt(const __bf16* __restrict__ A,
                                                    const __bf16* __restrict__ Bw,
                                                    void* __restrict__ Cout,
                                                    int M, int N, int K) {
    __shared__ __bf16 Bs[2][64 * 32];                // double-buffered B tile (8 KB)

    const int tid  = threadIdx.x;
    const int lane = tid & 31;
    const int wave = tid >> 5;
    const int half = lane >> 4, ln = lane & 15;
    const int m0 = blockIdx.y * 128 + wave * 16;
    const int n0 = blockIdx.x * 64;

    // per-thread staging slot: 64 rows x 32 bf16 = 256 x 16B chunks
    const int srow   = tid >> 2;                     // 0..63 (N row in tile)
    const int schunk = (tid & 3) * 8;                // bf16 offset in 32-wide K strip
    const __bf16* gB = Bw + (size_t)(n0 + srow) * K + schunk;

    async_copy_b128(&Bs[0][srow * 32 + schunk], gB); // prologue: stage tile 0

    v8f acc[4] = {};
    for (int k = 0; k < K; k += 32) {
        const int buf = (k >> 5) & 1;
        if (k + 32 < K) {
            async_copy_b128(&Bs[buf ^ 1][srow * 32 + schunk], gB + (k + 32));
            wait_async<1>();                         // current tile landed, next in flight
        } else {
            wait_async<0>();
        }
        __syncthreads();

        Frag a = load_frag_a(A, K, m0, k, lane);
        if (k + 32 < K)                              // global_prefetch_b8 for next A strip
            __builtin_prefetch(A + (size_t)(m0 + ln) * K + k + 32, 0, 1);
#pragma unroll
        for (int t = 0; t < 4; ++t) {
            Frag b = load_frag_b(Bs[buf], 32, t * 16, 0, lane);   // ds_load_b128 x2
            acc[t] = wmma_bf16(a.v, b.v, acc[t]);
        }
        __syncthreads();                             // protect buffer before re-staging
    }

#pragma unroll
    for (int t = 0; t < 4; ++t)
#pragma unroll
        for (int j = 0; j < 8; ++j) {
            const int row = m0 + j + (half << 3);
            const int col = n0 + t * 16 + ln;
            if constexpr (OUT_F32)
                ((float*)Cout)[(size_t)row * N + col] = acc[t][j];
            else
                ((__bf16*)Cout)[(size_t)row * N + col] = f2bf(acc[t][j]);
        }
}

// ---------------- RoPE in-place on Q and K (bf16, [B,S,NH,HD]) ----------------
__global__ __launch_bounds__(256) void rope_qk(__bf16* __restrict__ Q,
                                               __bf16* __restrict__ K) {
    int idx = blockIdx.x * 256 + threadIdx.x;      // over B*S*NH*(HD/2)
    const int d = idx & 63;
    const int h = (idx >> 6) & (NH_ - 1);
    const int s = (idx >> 10) & (S_ - 1);
    const int b = idx >> 21;
    if (b >= B_) return;

    // inv_freq = 10000^(-d/64); theta = s * inv_freq
    const float inv_freq = __expf(-(float)d * (9.210340371976184f / 64.0f));
    float sn, cs;
    __sincosf((float)s * inv_freq, &sn, &cs);

    const size_t base = ((size_t)(b * S_ + s)) * HID_ + h * HD_ + d;
    float q1 = bf2f(Q[base]), q2 = bf2f(Q[base + 64]);
    Q[base]      = f2bf(q1 * cs - q2 * sn);
    Q[base + 64] = f2bf(q2 * cs + q1 * sn);
    float k1 = bf2f(K[base]), k2 = bf2f(K[base + 64]);
    K[base]      = f2bf(k1 * cs - k2 * sn);
    K[base + 64] = f2bf(k2 * cs + k1 * sn);
}

// ---------------- V transpose: [B,S,NH,HD] -> [B,NH,HD,S] ----------------
__global__ __launch_bounds__(256) void transpose_v(const __bf16* __restrict__ Vraw,
                                                   __bf16* __restrict__ Vt) {
    int idx = blockIdx.x * 256 + threadIdx.x;      // over B*NH*HD*S, s fastest (coalesced writes)
    const int s = idx & (S_ - 1);
    const int d = (idx >> 11) & (HD_ - 1);
    const int h = (idx >> 18) & (NH_ - 1);
    const int b = idx >> 22;
    if (b >= B_) return;
    Vt[idx] = Vraw[((size_t)(b * S_ + s)) * HID_ + h * HD_ + d];
}

// ---------------- flash attention (causal, online softmax, all-WMMA) ----------------
// grid = (S/64, NH, B), block = 128 threads = 4 waves; each wave owns one 16-row Q tile.
__global__ __launch_bounds__(128) void flash_attn(const __bf16* __restrict__ Q,
                                                  const __bf16* __restrict__ K,
                                                  const __bf16* __restrict__ Vt,
                                                  __bf16* __restrict__ Oattn) {
    __shared__ __bf16 ldsP[4][16 * 32];            // per-wave P tile (C-layout -> A-layout staging)

    const int lane = threadIdx.x & 31;
    const int wave = threadIdx.x >> 5;
    const int half = lane >> 4, ln = lane & 15;
    const int b = blockIdx.z, h = blockIdx.y;
    const int qtile = blockIdx.x * 4 + wave;
    const int qbase = qtile * 16;

    const __bf16* Qb = Q  + (size_t)b * S_ * HID_ + h * HD_;
    const __bf16* Kb = K  + (size_t)b * S_ * HID_ + h * HD_;
    const __bf16* Vb = Vt + ((size_t)(b * NH_ + h)) * HD_ * S_;
    __bf16* P = ldsP[wave];

    // preload the four 16x32 Q fragments (K-dim = HD = 128)
    Frag qf[4];
#pragma unroll
    for (int kk = 0; kk < 4; ++kk) qf[kk] = load_frag_a(Qb, HID_, qbase, kk * 32, lane);

    float mrow[8], lrow[8];
    v8f o[8] = {};
#pragma unroll
    for (int j = 0; j < 8; ++j) { mrow[j] = -1e30f; lrow[j] = 0.f; }

    const int kmax = qbase + 16;                   // causal: keys <= last query row
    for (int kc = 0; kc < kmax; kc += 32) {
        // ---- S = Q * K^T for a 16x32 key chunk (two 16x16 tiles) ----
        v8f s0 = {}, s1 = {};
#pragma unroll
        for (int kk = 0; kk < 4; ++kk) {
            Frag b0 = load_frag_b(Kb, HID_, kc,      kk * 32, lane);
            Frag b1 = load_frag_b(Kb, HID_, kc + 16, kk * 32, lane);
            s0 = wmma_bf16(qf[kk].v, b0.v, s0);
            s1 = wmma_bf16(qf[kk].v, b1.v, s1);
        }
        // ---- online softmax (row reductions across 16-lane halves, wave32) ----
#pragma unroll
        for (int j = 0; j < 8; ++j) {
            const int row = qbase + j + (half << 3);
            const int c0 = kc + ln, c1 = c0 + 16;
            float v0 = (c0 <= row) ? s0[j] * SCALE_ : -1e30f;
            float v1 = (c1 <= row) ? s1[j] * SCALE_ : -1e30f;
            float mx = fmaxf(v0, v1);
#pragma unroll
            for (int off = 8; off; off >>= 1) mx = fmaxf(mx, __shfl_xor(mx, off, 16));
            const float mnew  = fmaxf(mrow[j], mx);
            const float alpha = __expf(mrow[j] - mnew);
            const float p0 = __expf(v0 - mnew);
            const float p1 = __expf(v1 - mnew);
            float rs = p0 + p1;
#pragma unroll
            for (int off = 8; off; off >>= 1) rs += __shfl_xor(rs, off, 16);
            lrow[j] = lrow[j] * alpha + rs;
            mrow[j] = mnew;
#pragma unroll
            for (int dt = 0; dt < 8; ++dt) o[dt][j] *= alpha;   // rescale accumulators
            const int prow = j + (half << 3);
            P[prow * 32 + ln]      = f2bf(p0);
            P[prow * 32 + 16 + ln] = f2bf(p1);
        }
        // ---- O += P * V  (P via LDS in A-layout; V^T gives contiguous B-frags) ----
        Frag pf = load_frag_a(P, 32, 0, 0, lane);  // ds_load_b128 x2; DScnt RAW handled by compiler
#pragma unroll
        for (int dt = 0; dt < 8; ++dt) {
            Frag vf = load_frag_b(Vb, S_, dt * 16, kc, lane);
            o[dt] = wmma_bf16(pf.v, vf.v, o[dt]);
        }
    }

    // ---- epilogue: O /= l, write bf16 [B,S,HID] ----
#pragma unroll
    for (int j = 0; j < 8; ++j) {
        const float inv = 1.0f / lrow[j];
        const int row = qbase + j + (half << 3);
        __bf16* outp = Oattn + ((size_t)(b * S_ + row)) * HID_ + h * HD_;
#pragma unroll
        for (int dt = 0; dt < 8; ++dt)
            outp[dt * 16 + ln] = f2bf(o[dt][j] * inv);
    }
}

// ---------------- host-side orchestration ----------------
extern "C" void kernel_launch(void* const* d_in, const int* in_sizes, int n_in,
                              void* d_out, int out_size, void* d_ws, size_t ws_size,
                              hipStream_t stream) {
    const float* hs = (const float*)d_in[0];
    // d_in[1] attention_mask (causal; computed analytically), d_in[2] position_ids (= arange)
    const float* Wq = (const float*)d_in[3];
    const float* Wk = (const float*)d_in[4];
    const float* Wv = (const float*)d_in[5];
    const float* Wo = (const float*)d_in[6];

    const int M = B_ * S_;                          // 4096 token rows

    char* w = (char*)d_ws;
    __bf16* Xb  = (__bf16*)w; w += (size_t)M * HID_ * 2;
    __bf16* Wqb = (__bf16*)w; w += (size_t)HID_ * HID_ * 2;
    __bf16* Wkb = (__bf16*)w; w += (size_t)HID_ * HID_ * 2;
    __bf16* Wvb = (__bf16*)w; w += (size_t)HID_ * HID_ * 2;
    __bf16* Wob = (__bf16*)w; w += (size_t)HID_ * HID_ * 2;
    __bf16* Qr  = (__bf16*)w; w += (size_t)M * HID_ * 2;
    __bf16* Kr  = (__bf16*)w; w += (size_t)M * HID_ * 2;
    __bf16* Vr  = (__bf16*)w; w += (size_t)M * HID_ * 2;
    __bf16* Vt  = (__bf16*)w; w += (size_t)M * HID_ * 2;
    __bf16* At  = (__bf16*)w; w += (size_t)M * HID_ * 2;

    auto cvt = [&](const float* src, __bf16* dst, int n) {
        cvt_f32_bf16<<<dim3((n / 4 + 255) / 256), dim3(256), 0, stream>>>(src, dst, n);
    };
    cvt(hs, Xb, M * HID_);
    cvt(Wq, Wqb, HID_ * HID_);
    cvt(Wk, Wkb, HID_ * HID_);
    cvt(Wv, Wvb, HID_ * HID_);
    cvt(Wo, Wob, HID_ * HID_);

    const dim3 gemm_grid(HID_ / 64, M / 128), gemm_blk(256);
    gemm_bf16_nt<false><<<gemm_grid, gemm_blk, 0, stream>>>(Xb, Wqb, Qr, M, HID_, HID_);
    gemm_bf16_nt<false><<<gemm_grid, gemm_blk, 0, stream>>>(Xb, Wkb, Kr, M, HID_, HID_);
    gemm_bf16_nt<false><<<gemm_grid, gemm_blk, 0, stream>>>(Xb, Wvb, Vr, M, HID_, HID_);

    rope_qk<<<dim3((B_ * S_ * NH_ * 64) / 256), dim3(256), 0, stream>>>(Qr, Kr);
    transpose_v<<<dim3((B_ * NH_ * HD_ * S_) / 256), dim3(256), 0, stream>>>(Vr, Vt);

    flash_attn<<<dim3(S_ / 64, NH_, B_), dim3(128), 0, stream>>>(Qr, Kr, Vt, At);

    gemm_bf16_nt<true><<<gemm_grid, gemm_blk, 0, stream>>>(At, Wob, d_out, M, HID_, HID_);
}
